// Multihead_Self_Attention_83854941487151
// MI455X (gfx1250) — compile-verified, matrix operand order checked by reading, not run on hardware
//
#include <hip/hip_runtime.h>

#define B_  4
#define S_  2048
#define D_  1024
#define H_  16
#define DK_ 64
#define LN_THETA 9.210340371976184f  // ln(10000)

typedef __attribute__((ext_vector_type(16))) __bf16 v16bf;
typedef __attribute__((ext_vector_type(8)))  __bf16 v8bf;
typedef __attribute__((ext_vector_type(4)))  __bf16 v4bf;
typedef __attribute__((ext_vector_type(8)))  float  v8f;
typedef __attribute__((ext_vector_type(4)))  float  v4f;

static __device__ __forceinline__ v16bf cat16(v8bf lo, v8bf hi) {
  return __builtin_shufflevector(lo, hi, 0,1,2,3,4,5,6,7,8,9,10,11,12,13,14,15);
}

static __device__ __forceinline__ v8f wmma_bf16(v16bf a, v16bf b, v8f c) {
  // D = A(16x32) * B(32x16) + C, fp32 accumulate
  return __builtin_amdgcn_wmma_f32_16x16x32_bf16(false, a, false, b, (short)0, c,
                                                 false, false);
}

// CDNA5 async global->LDS copy, 16 bytes per lane. Tracked by ASYNCcnt.
static __device__ __forceinline__ void async_cp_b128(void* lds_ptr, const void* gaddr) {
  unsigned loff = (unsigned)(__UINTPTR_TYPE__)(__attribute__((address_space(3))) char*)lds_ptr;
  asm volatile("global_load_async_to_lds_b128 %0, %1, off"
               :: "v"(loff), "v"(gaddr)
               : "memory");
}

static __device__ __forceinline__ void wait_async0() {
  asm volatile("s_wait_asynccnt 0x0" ::: "memory");
}

// ---------------------------------------------------------------------------
// fp32 -> bf16 conversion (vectorized x4)
// ---------------------------------------------------------------------------
__global__ __launch_bounds__(256) void f32_to_bf16(const float* __restrict__ in,
                                                   __bf16* __restrict__ out,
                                                   int n4) {
  int i = blockIdx.x * 256 + threadIdx.x;
  if (i >= n4) return;
  v4f v = ((const v4f*)in)[i];
  v4bf o;
  o[0] = (__bf16)v[0]; o[1] = (__bf16)v[1];
  o[2] = (__bf16)v[2]; o[3] = (__bf16)v[3];
  ((v4bf*)out)[i] = o;
}

// ---------------------------------------------------------------------------
// Y = A[8192,1024] * W[1024,1024]^T via bf16 WMMA, fp32 accumulate.
// Block = 256 thr = 8 waves; block tile 256(M) x 64(N); wave tile 32 x 64.
// W k-chunk (64x32 bf16 = 4KB) staged into double-buffered LDS with
// global_load_async_to_lds_b128; async copy of chunk k+1 overlaps WMMA of k.
// mode 0: Q  -> RoPE, store bf16 [B,H,S,DK]
// mode 1: K  -> RoPE, store bf16 [B,H,S,DK]
// mode 2: V  ->        store bf16 [B,H,DK,S]
// mode 3: O-proj ->    store fp32 [B*S, D]
// ---------------------------------------------------------------------------
__global__ __launch_bounds__(256) void gemm_bf16(
    const __bf16* __restrict__ A, const __bf16* __restrict__ W,
    const int* __restrict__ pos, __bf16* __restrict__ outB,
    float* __restrict__ outF, int mode) {
  __shared__ __bf16 wtile[2][64 * 32];  // [buf][n(64) x k(32)]
  const int tid  = threadIdx.x;
  const int lane = tid & 31;
  const int wave = tid >> 5;
  const int r    = lane & 15;
  const int half = lane >> 4;
  const int n0   = blockIdx.x * 64;
  const int m0   = blockIdx.y * 256 + wave * 32;

  // async W staging: thread copies 16B; row = tid/4 (64 rows), seg = tid%4
  const int wrow = tid >> 2, wseg = tid & 3;
  const __bf16* wsrc = W + (size_t)(n0 + wrow) * D_ + wseg * 8;
  const int wdst = wrow * 32 + wseg * 8;  // element offset within buffer

  // prologue: stage k-chunk 0 into buffer 0
  async_cp_b128(&wtile[0][wdst], wsrc);

  v8f acc[2][4] = {};
  const __bf16* arow0 = A + (size_t)(m0 + r) * D_;
  const __bf16* arow1 = A + (size_t)(m0 + 16 + r) * D_;

  for (int kc = 0; kc < D_ / 32; ++kc) {
    const int k0  = kc * 32;
    const int cur = kc & 1;
    wait_async0();        // this wave's copies into buf `cur` done
    __syncthreads();      // all waves' copies done; all readers of buf cur^1 done
    if (kc + 1 < D_ / 32)
      async_cp_b128(&wtile[cur ^ 1][wdst], wsrc + (k0 + 32));

    // ---- load all fragments first (distinct registers), then 8 WMMA ----
    v16bf a0 = cat16(*(const v8bf*)(arow0 + k0 + half * 8),
                     *(const v8bf*)(arow0 + k0 + 16 + half * 8));
    v16bf a1 = cat16(*(const v8bf*)(arow1 + k0 + half * 8),
                     *(const v8bf*)(arow1 + k0 + 16 + half * 8));
    v16bf bf[4];
#pragma unroll
    for (int t = 0; t < 4; ++t) {
      const __bf16* wr_ = &wtile[cur][(t * 16 + r) * 32 + half * 8];
      bf[t] = cat16(*(const v8bf*)wr_, *(const v8bf*)(wr_ + 16));
    }
#pragma unroll
    for (int t = 0; t < 4; ++t) acc[0][t] = wmma_bf16(a0, bf[t], acc[0][t]);
#pragma unroll
    for (int t = 0; t < 4; ++t) acc[1][t] = wmma_bf16(a1, bf[t], acc[1][t]);
  }

  // Epilogue: C layout = (row j + 8*half, col = n0 + t*16 + r)
#pragma unroll
  for (int mi = 0; mi < 2; ++mi) {
#pragma unroll
    for (int t = 0; t < 4; ++t) {
      const int col = n0 + t * 16 + r;
#pragma unroll
      for (int j = 0; j < 8; ++j) {
        const int g = m0 + mi * 16 + j + 8 * half;  // row in [0, B*S)
        float v = acc[mi][t][j];
        if (mode <= 1) {
          // RoPE: pairs (2i, 2i+1) live in adjacent lanes of the C layout
          const int dk = col & (DK_ - 1);
          const int s  = g & (S_ - 1);
          const float p   = (float)pos[s];
          const float inv = __expf(-(float)(dk & ~1) * (LN_THETA / (float)DK_));
          float sn, cs;
          __sincosf(p * inv, &sn, &cs);
          const float other = __shfl_xor(v, 1, 32);
          v = (col & 1) ? fmaf(v, cs, other * sn)    // x1*sin + x2*cos
                        : fmaf(v, cs, -other * sn);  // x1*cos - x2*sin
        }
        if (mode == 3) {
          outF[(size_t)g * D_ + col] = v;
        } else {
          const int b  = g >> 11;
          const int s  = g & (S_ - 1);
          const int h  = col >> 6;
          const int dk = col & (DK_ - 1);
          size_t idx;
          if (mode == 2) idx = ((size_t)(b * H_ + h) * DK_ + dk) * S_ + s;
          else           idx = ((size_t)(b * H_ + h) * S_ + s) * DK_ + dk;
          outB[idx] = (__bf16)v;
        }
      }
    }
  }
}

// ---------------------------------------------------------------------------
// Causal flash attention: one wave per 16-query tile, 32-wide KV chunks,
// online softmax in fp32, QK^T and P*V on bf16 WMMA. P transposed C->A
// layout through per-wave LDS staging (ds_store_b16 -> s_wait_dscnt ->
// ds_load_b128). Fragments grouped before WMMA for load/math overlap.
// ---------------------------------------------------------------------------
__global__ __launch_bounds__(256) void flash_attn(
    const __bf16* __restrict__ Q,    // [B,H,S,DK]
    const __bf16* __restrict__ K,    // [B,H,S,DK]
    const __bf16* __restrict__ Vt,   // [B,H,DK,S]
    __bf16* __restrict__ O) {        // [B,S,H*DK] bf16
  __shared__ __bf16 plds[8 * 16 * 32];
  const int lane = threadIdx.x & 31;
  const int wave = threadIdx.x >> 5;
  const int r    = lane & 15;
  const int half = lane >> 4;
  const int b  = blockIdx.z;
  const int h  = blockIdx.y;
  const int q0 = (blockIdx.x * 8 + wave) * 16;

  const __bf16* Qb = Q  + (size_t)(b * H_ + h) * S_ * DK_;
  const __bf16* Kb = K  + (size_t)(b * H_ + h) * S_ * DK_;
  const __bf16* Vb = Vt + (size_t)(b * H_ + h) * DK_ * S_;
  __bf16* lds = plds + wave * (16 * 32);

  // Preload Q as two A-fragments (d = 0..31, 32..63)
  v16bf qf[2];
#pragma unroll
  for (int c = 0; c < 2; ++c) {
    const __bf16* qr = Qb + (size_t)(q0 + r) * DK_ + c * 32 + half * 8;
    qf[c] = cat16(*(const v8bf*)qr, *(const v8bf*)(qr + 16));
  }

  float m_i[8], l_i[8];
  v8f o_acc[4] = {};
#pragma unroll
  for (int j = 0; j < 8; ++j) { m_i[j] = -1e30f; l_i[j] = 0.f; }

  const int nchunk = (q0 + 47) >> 5;  // cover kv <= q0+15
  for (int kc = 0; kc < nchunk; ++kc) {
    const int kv0 = kc * 32;
    if (kc + 1 < nchunk)
      __builtin_prefetch(Kb + (size_t)(kv0 + 32 + r) * DK_, 0, 1);

    // ---- load all 4 K fragments, then 4 score WMMAs ----
    v16bf kf[2][2];
#pragma unroll
    for (int t = 0; t < 2; ++t)
#pragma unroll
      for (int c = 0; c < 2; ++c) {
        const __bf16* kr =
            Kb + (size_t)(kv0 + t * 16 + r) * DK_ + c * 32 + half * 8;
        kf[t][c] = cat16(*(const v8bf*)kr, *(const v8bf*)(kr + 16));
      }
    v8f sc[2] = {};
#pragma unroll
    for (int t = 0; t < 2; ++t)
#pragma unroll
      for (int c = 0; c < 2; ++c) sc[t] = wmma_bf16(qf[c], kf[t][c], sc[t]);

    // 1/sqrt(DK) scale + causal mask (select, no EXEC divergence)
#pragma unroll
    for (int t = 0; t < 2; ++t) {
      const int col = kv0 + t * 16 + r;
#pragma unroll
      for (int j = 0; j < 8; ++j) {
        const int row = q0 + j + 8 * half;
        sc[t][j] = (col <= row) ? sc[t][j] * 0.125f : -1e30f;
      }
    }

    // online softmax: rowwise max/sum = 16-lane shfl_xor (stays within half)
    float alpha[8];
#pragma unroll
    for (int j = 0; j < 8; ++j) {
      float mx = fmaxf(sc[0][j], sc[1][j]);
#pragma unroll
      for (int d = 1; d < 16; d <<= 1) mx = fmaxf(mx, __shfl_xor(mx, d, 32));
      const float mn = fmaxf(m_i[j], mx);
      alpha[j] = __expf(m_i[j] - mn);
      m_i[j]   = mn;
      const float p0 = __expf(sc[0][j] - mn);
      const float p1 = __expf(sc[1][j] - mn);
      sc[0][j] = p0; sc[1][j] = p1;
      float rs = p0 + p1;
#pragma unroll
      for (int d = 1; d < 16; d <<= 1) rs += __shfl_xor(rs, d, 32);
      l_i[j] = alpha[j] * l_i[j] + rs;
    }
#pragma unroll
    for (int dt = 0; dt < 4; ++dt)
#pragma unroll
      for (int j = 0; j < 8; ++j) o_acc[dt][j] *= alpha[j];

    // stage P: C layout -> LDS row-major 16x32 bf16 (per-wave region)
#pragma unroll
    for (int t = 0; t < 2; ++t)
#pragma unroll
      for (int j = 0; j < 8; ++j)
        lds[(j + 8 * half) * 32 + t * 16 + r] = (__bf16)sc[t][j];

    // V fragment loads (global) overlap the LDS round-trip
    v16bf vf[4];
#pragma unroll
    for (int dt = 0; dt < 4; ++dt) {
      const __bf16* vr = Vb + (size_t)(dt * 16 + r) * S_ + kv0 + half * 8;
      vf[dt] = cat16(*(const v8bf*)vr, *(const v8bf*)(vr + 16));
    }

    asm volatile("s_wait_dscnt 0x0" ::: "memory");

    // reload P as an A-fragment (K = kv within chunk, 32 wide)
    const __bf16* pr = lds + r * 32 + half * 8;
    v16bf pf = cat16(*(const v8bf*)pr, *(const v8bf*)(pr + 16));

#pragma unroll
    for (int dt = 0; dt < 4; ++dt) o_acc[dt] = wmma_bf16(pf, vf[dt], o_acc[dt]);
  }

  // finalize 1/l and store bf16 [B,S,H*DK]
#pragma unroll
  for (int j = 0; j < 8; ++j) {
    const float invl = 1.0f / l_i[j];
    const int row = q0 + j + 8 * half;
#pragma unroll
    for (int dt = 0; dt < 4; ++dt)
      O[(size_t)(b * S_ + row) * D_ + h * DK_ + dt * 16 + r] =
          (__bf16)(o_acc[dt][j] * invl);
  }
}

// ---------------------------------------------------------------------------
extern "C" void kernel_launch(void* const* d_in, const int* in_sizes, int n_in,
                              void* d_out, int out_size, void* d_ws, size_t ws_size,
                              hipStream_t stream) {
  const float* x  = (const float*)d_in[0];
  const float* Wq = (const float*)d_in[1];
  const float* Wk = (const float*)d_in[2];
  const float* Wv = (const float*)d_in[3];
  const float* Wo = (const float*)d_in[4];
  const int*  pos = (const int*)d_in[5];
  float* out = (float*)d_out;

  char* ws = (char*)d_ws;
  size_t off = 0;
  auto alloc = [&](size_t bytes) -> char* {
    char* p = ws + off;
    off += (bytes + 255) & ~(size_t)255;
    return p;
  };

  const size_t nX = (size_t)B_ * S_ * D_;   // 8,388,608
  const size_t nW = (size_t)D_ * D_;        // 1,048,576

  __bf16* xb  = (__bf16*)alloc(nX * 2);
  __bf16* wqb = (__bf16*)alloc(nW * 2);
  __bf16* wkb = (__bf16*)alloc(nW * 2);
  __bf16* wvb = (__bf16*)alloc(nW * 2);
  __bf16* wob = (__bf16*)alloc(nW * 2);
  __bf16* Qb  = (__bf16*)alloc(nX * 2);     // [B,H,S,DK]
  __bf16* Kb  = (__bf16*)alloc(nX * 2);     // [B,H,S,DK]
  __bf16* Vtb = (__bf16*)alloc(nX * 2);     // [B,H,DK,S]
  __bf16* Ab  = (__bf16*)alloc(nX * 2);     // attention out [B,S,D]

  // 1) fp32 -> bf16 conversions
  {
    int n4 = (int)(nX / 4);
    f32_to_bf16<<<(n4 + 255) / 256, 256, 0, stream>>>(x, xb, n4);
    int w4 = (int)(nW / 4);
    f32_to_bf16<<<(w4 + 255) / 256, 256, 0, stream>>>(Wq, wqb, w4);
    f32_to_bf16<<<(w4 + 255) / 256, 256, 0, stream>>>(Wk, wkb, w4);
    f32_to_bf16<<<(w4 + 255) / 256, 256, 0, stream>>>(Wv, wvb, w4);
    f32_to_bf16<<<(w4 + 255) / 256, 256, 0, stream>>>(Wo, wob, w4);
  }

  // 2) fused projection GEMMs (+RoPE for Q/K, transpose-store for V)
  dim3 gg(D_ / 64, (B_ * S_) / 256);
  gemm_bf16<<<gg, 256, 0, stream>>>(xb, wqb, pos, Qb,  nullptr, 0);
  gemm_bf16<<<gg, 256, 0, stream>>>(xb, wkb, pos, Kb,  nullptr, 1);
  gemm_bf16<<<gg, 256, 0, stream>>>(xb, wvb, pos, Vtb, nullptr, 2);

  // 3) causal flash attention
  dim3 ga(S_ / (16 * 8), H_, B_);
  flash_attn<<<ga, 256, 0, stream>>>(Qb, Kb, Vtb, Ab);

  // 4) output projection -> fp32 d_out
  gemm_bf16<<<gg, 256, 0, stream>>>(Ab, wob, pos, nullptr, out, 3);
}
